// TaskConditionedAdapter_81441169867239
// MI455X (gfx1250) — matrix operand order repaired and checked by reading, not compiled, not run on hardware
//
#include <hip/hip_runtime.h>
#include <hip/hip_bf16.h>
#include <math.h>

// ---------------- problem constants (from reference setup_inputs) ----------
#define Bn   16
#define Hn   64
#define Wn   64
#define Cn   768
#define HWn  (Hn*Wn)          // 4096
#define Rn   64
#define Tn   8
#define NPS  (HWn*Cn)         // 3,145,728 elements per sample
#define PB   64               // partial-reduction blocks per sample
#define CHUNK (NPS/PB)        // 49,152 elements per partial block
#define MT   32               // rows (of HW) per main-kernel block
#define LDA  (Cn + 8)         // padded ushort row stride for normalized-x LDS
#define LDH  (Rn + 8)         // padded ushort row stride for h LDS

#define KT1  (Cn/32)          // 24 k-tiles for GEMM1
#define NT1  (Rn/16)          // 4  n-tiles for GEMM1
#define NT2  (Cn/16)          // 48 n-tiles for GEMM2
#define KT2  (Rn/32)          // 2  k-tiles for GEMM2
#define TILE_ELEMS 512        // 32 lanes * 16 bf16 per fragment tile
#define WD_PACK_PER_TASK (KT1*NT1*TILE_ELEMS)  // 49152 ushorts
#define WU_PACK_PER_TASK (NT2*KT2*TILE_ELEMS)  // 49152 ushorts

typedef __attribute__((ext_vector_type(16))) __bf16 v16bf;
typedef __attribute__((ext_vector_type(8)))  float  v8f;

union Frag {
    v16bf v;
    uint4 q[2];
    unsigned short u[16];
};

// float -> bf16 bits, round-to-nearest-even
__device__ __forceinline__ unsigned short f2bf(float f) {
    unsigned int u = __float_as_uint(f);
    unsigned int rnd = 0x7FFFu + ((u >> 16) & 1u);
    return (unsigned short)((u + rnd) >> 16);
}

// -------------------- kernel 1: per-sample partial sums --------------------
__global__ __launch_bounds__(256)
void stats_partial(const float* __restrict__ x, float* __restrict__ partial) {
    const int blk = blockIdx.x;
    const int b = blk / PB, p = blk % PB;
    const float4* base = (const float4*)(x + (size_t)b * NPS + (size_t)p * CHUNK);
    const int nvec = CHUNK / 4;           // 12288 float4
    float s = 0.f, s2 = 0.f;
    for (int i = threadIdx.x; i < nvec; i += 256) {
        float4 v = base[i];
        s  += v.x + v.y + v.z + v.w;
        s2 += v.x*v.x + v.y*v.y + v.z*v.z + v.w*v.w;
    }
    __shared__ float sh[256], sh2[256];
    sh[threadIdx.x] = s; sh2[threadIdx.x] = s2;
    __syncthreads();
    for (int off = 128; off > 0; off >>= 1) {
        if (threadIdx.x < off) {
            sh[threadIdx.x]  += sh[threadIdx.x + off];
            sh2[threadIdx.x] += sh2[threadIdx.x + off];
        }
        __syncthreads();
    }
    if (threadIdx.x == 0) {
        partial[blk*2]   = sh[0];
        partial[blk*2+1] = sh2[0];
    }
}

// -------------------- kernel 2: finalize mean / inv_std --------------------
__global__ __launch_bounds__(64)
void stats_final(const float* __restrict__ partial, float* __restrict__ stats) {
    const int b = blockIdx.x;
    const int t = threadIdx.x;
    __shared__ float sh[64], sh2[64];
    sh[t]  = partial[(b*PB + t)*2];
    sh2[t] = partial[(b*PB + t)*2 + 1];
    __syncthreads();
    for (int off = 32; off > 0; off >>= 1) {
        if (t < off) { sh[t] += sh[t+off]; sh2[t] += sh2[t+off]; }
        __syncthreads();
    }
    if (t == 0) {
        float mean = sh[0] / (float)NPS;
        float var  = sh2[0] / (float)NPS - mean*mean;
        stats[b*2]     = mean;
        stats[b*2 + 1] = rsqrtf(var + 1e-5f);
    }
}

// ---------- kernel 2b: pre-pack W_down into bf16 WMMA-B lane order ---------
// B 32x16 bf16 fragment: lane n = lane&15, K = (e>>3)*16 + (lane>>4)*8 + (e&7) (table order).
// Tile order: (((task*KT1 + kt)*NT1 + nt)*512 + lane*16 + e)
__global__ __launch_bounds__(256)
void pack_wd(const float* __restrict__ Wd, unsigned short* __restrict__ pd) {
    int o = blockIdx.x * 256 + threadIdx.x;            // < 8*49152
    int t = o / WD_PACK_PER_TASK;
    int rem = o - t * WD_PACK_PER_TASK;
    int tile = rem / TILE_ELEMS;                       // kt*NT1 + nt
    int kt = tile / NT1, nt = tile - kt * NT1;
    int r = rem - tile * TILE_ELEMS;
    int lane = r >> 4, e = r & 15;
    int n = nt*16 + (lane & 15);
    int k = kt*32 + (e >> 3)*16 + (lane >> 4)*8 + (e & 7);
    pd[o] = f2bf(Wd[((size_t)t * Rn + n) * Cn + k]);
}

// ---------- kernel 2c: pre-pack W_up into bf16 WMMA-B lane order -----------
// Tile order: (((task*NT2 + nt)*KT2 + kt)*512 + lane*16 + e)
__global__ __launch_bounds__(256)
void pack_wu(const float* __restrict__ Wu, unsigned short* __restrict__ pu) {
    int o = blockIdx.x * 256 + threadIdx.x;            // < 8*49152
    int t = o / WU_PACK_PER_TASK;
    int rem = o - t * WU_PACK_PER_TASK;
    int tile = rem / TILE_ELEMS;                       // nt*KT2 + kt
    int nt = tile / KT2, kt = tile - nt * KT2;
    int r = rem - tile * TILE_ELEMS;
    int lane = r >> 4, e = r & 15;
    int n = nt*16 + (lane & 15);
    int k = kt*32 + (e >> 3)*16 + (lane >> 4)*8 + (e & 7);
    pu[o] = f2bf(Wu[((size_t)t * Cn + n) * Rn + k]);
}

// -------------------- kernel 3: fused norm + GEMM1 + GELU + GEMM2 + residual
__global__ __launch_bounds__(256)
void adapter_main(const float* __restrict__ x, const int* __restrict__ task_ids,
                  const float* __restrict__ gamma, const float* __restrict__ beta,
                  const unsigned short* __restrict__ pwd,
                  const unsigned short* __restrict__ pwu,
                  const float* __restrict__ scales, const float* __restrict__ stats,
                  float* __restrict__ out) {
    __shared__ __align__(16) unsigned short As[MT * LDA]; // normalized x, bf16
    __shared__ __align__(16) unsigned short Hs[MT * LDH]; // gelu(h), bf16

    const int b    = blockIdx.y;
    const int row0 = blockIdx.x * MT;     // base spatial index (hw)
    const int tid  = threadIdx.x;
    const int lane = tid & 31;
    const int wv   = tid >> 5;            // wave id 0..7
    const int ln   = lane & 15;
    const int hi   = lane >> 4;           // half-wave select

    const int   task = task_ids[b];
    const float s    = scales[task];
    const float mean = stats[b*2];
    const float inv  = stats[b*2 + 1];

    const float* xb = x   + ((size_t)b * HWn + row0) * Cn;
    float*       ob = out + ((size_t)b * HWn + row0) * Cn;
    const unsigned short* wd = pwd + (size_t)task * WD_PACK_PER_TASK;
    const unsigned short* wu = pwu + (size_t)task * WU_PACK_PER_TASK;

    // ---- Stage 1: GroupNorm affine into bf16 LDS tile (coalesced float4) --
    for (int i = tid; i < (MT * Cn) / 4; i += 256) {
        int lin = i * 4;
        int r = lin / Cn;
        int c = lin - r * Cn;
        float4 v  = *(const float4*)(xb + (size_t)r * Cn + c);
        float4 g  = *(const float4*)(gamma + c);
        float4 be = *(const float4*)(beta + c);
        unsigned short* dst = &As[r * LDA + c];
        dst[0] = f2bf((v.x - mean) * inv * g.x + be.x);
        dst[1] = f2bf((v.y - mean) * inv * g.y + be.y);
        dst[2] = f2bf((v.z - mean) * inv * g.z + be.z);
        dst[3] = f2bf((v.w - mean) * inv * g.w + be.w);
    }
    __syncthreads();

    // ---- Stage 2: h = x_norm · Wd^T  (M=32, N=64, K=768), one 16x16/wave --
    {
        const int m0 = (wv >> 2) * 16;    // 0 or 16
        const int n0t = (wv & 3);         // n-tile 0..3
        v8f acc = {0.f,0.f,0.f,0.f,0.f,0.f,0.f,0.f};
        for (int kt = 0; kt < KT1; kt++) {
            Frag a, bb;
            // A 16x32 bf16 from LDS: two contiguous 16B chunks per lane
            const unsigned short* ap = &As[(m0 + ln) * LDA + kt*32 + hi*8];
            a.q[0] = *(const uint4*)ap;
            a.q[1] = *(const uint4*)(ap + 16);
            // B fragment: pre-packed, lane-contiguous 32B (coalesced 1KB/wave)
            const unsigned short* bp = wd + ((size_t)kt * NT1 + n0t) * TILE_ELEMS + lane*16;
            bb.q[0] = *(const uint4*)bp;
            bb.q[1] = *(const uint4*)(bp + 8);
            acc = __builtin_amdgcn_wmma_f32_16x16x32_bf16(
                      false, a.v, false, bb.v, (short)0, acc, false, false);
        }
        // exact-erf GELU, store bf16 h tile
        #pragma unroll
        for (int r = 0; r < 8; r++) {
            float v  = acc[r];
            float gl = 0.5f * v * (1.0f + erff(v * 0.70710678118f));
            Hs[(m0 + hi*8 + r) * LDH + (n0t*16 + ln)] = f2bf(gl);
        }
    }
    __syncthreads();

    // ---- Stage 3: delta = gelu(h) · Wu^T (M=32, N=768, K=64) + residual ---
    // 48 N-tiles / 8 waves = 6 panels per wave; B frags reused across both M-tiles
    for (int j = 0; j < 6; j++) {
        const int nt = wv * 6 + j;
        Frag b0, b1;
        const unsigned short* bp0 = wu + ((size_t)nt * KT2 + 0) * TILE_ELEMS + lane*16;
        const unsigned short* bp1 = wu + ((size_t)nt * KT2 + 1) * TILE_ELEMS + lane*16;
        b0.q[0] = *(const uint4*)bp0;
        b0.q[1] = *(const uint4*)(bp0 + 8);
        b1.q[0] = *(const uint4*)bp1;
        b1.q[1] = *(const uint4*)(bp1 + 8);

        #pragma unroll
        for (int mi = 0; mi < 2; mi++) {
            const int m0 = mi * 16;
            Frag a0, a1;
            const unsigned short* ap = &Hs[(m0 + ln) * LDH + hi*8];
            a0.q[0] = *(const uint4*)ap;          // K chunk 0..31
            a0.q[1] = *(const uint4*)(ap + 16);
            a1.q[0] = *(const uint4*)(ap + 32);   // K chunk 32..63
            a1.q[1] = *(const uint4*)(ap + 48);
            v8f acc = {0.f,0.f,0.f,0.f,0.f,0.f,0.f,0.f};
            acc = __builtin_amdgcn_wmma_f32_16x16x32_bf16(
                      false, a0.v, false, b0.v, (short)0, acc, false, false);
            acc = __builtin_amdgcn_wmma_f32_16x16x32_bf16(
                      false, a1.v, false, b1.v, (short)0, acc, false, false);
            // fused residual epilogue: out = x + s*delta
            // D layout: lane col = nt*16+ln, rows m0 + hi*8 + r
            #pragma unroll
            for (int r = 0; r < 8; r++) {
                int row = m0 + hi*8 + r;
                size_t idx = (size_t)row * Cn + (nt*16 + ln);
                ob[idx] = xb[idx] + s * acc[r];
            }
        }
    }
}

// ---------------------------------------------------------------------------
extern "C" void kernel_launch(void* const* d_in, const int* in_sizes, int n_in,
                              void* d_out, int out_size, void* d_ws, size_t ws_size,
                              hipStream_t stream) {
    (void)in_sizes; (void)n_in; (void)out_size; (void)ws_size;
    const float* x        = (const float*)d_in[0];
    const int*   task_ids = (const int*)  d_in[1];
    const float* gamma    = (const float*)d_in[2];
    const float* beta     = (const float*)d_in[3];
    const float* Wd       = (const float*)d_in[4];
    const float* Wu       = (const float*)d_in[5];
    const float* scales   = (const float*)d_in[6];
    float* out = (float*)d_out;

    // workspace layout
    float* partial = (float*)d_ws;                       // Bn*PB*2 floats (8 KB)
    float* statsp  = partial + Bn * PB * 2;              // 2*Bn floats
    unsigned short* pwd = (unsigned short*)(statsp + 2*Bn + 6); // align up
    pwd = (unsigned short*)(((uintptr_t)pwd + 15) & ~(uintptr_t)15);
    unsigned short* pwu = pwd + (size_t)Tn * WD_PACK_PER_TASK;

    stats_partial<<<Bn * PB, 256, 0, stream>>>(x, partial);
    stats_final<<<Bn, 64, 0, stream>>>(partial, statsp);
    pack_wd<<<(Tn * WD_PACK_PER_TASK) / 256, 256, 0, stream>>>(Wd, pwd);
    pack_wu<<<(Tn * WU_PACK_PER_TASK) / 256, 256, 0, stream>>>(Wu, pwu);

    dim3 grid(HWn / MT, Bn);
    adapter_main<<<grid, 256, 0, stream>>>(x, task_ids, gamma, beta,
                                           pwd, pwu, scales, statsp, out);
}
